// ReadoutLayer_16088947491016
// MI455X (gfx1250) — matrix operand order repaired and checked by reading, not compiled, not run on hardware
//
#include <hip/hip_runtime.h>
#include <hip/hip_bf16.h>

typedef __attribute__((ext_vector_type(16))) __bf16 v16bf;
typedef __attribute__((ext_vector_type(8)))  float  v8f;

#define BN_EPS 1e-3f

// ---------------------------------------------------------------------------
// Kernel 0: zero the pooled accumulator [G*64] floats in workspace
// ---------------------------------------------------------------------------
__global__ __launch_bounds__(256) void zero_pooled(float* __restrict__ pooled, int n) {
    int i = blockIdx.x * 256 + threadIdx.x;
    if (i < n) pooled[i] = 0.0f;
}

// ---------------------------------------------------------------------------
// Kernel 1: pack W_i [128,64] and W_j [64,64] (row-major f32) into bf16 hi/lo
// B-fragments in the contiguous k order node_kernel also uses for A:
//   fragment index f = (kc*4 + ct)  (kc = K/32 chunk, ct = N/16 tile)
//   lane L holds column N = ct*16 + (L%16), elements e=0..15 are
//   K = kc*32 + (L/16)*16 + e.
// One thread per (f, lane). 512 threads cover W_i; threads <256 also do W_j.
// ---------------------------------------------------------------------------
__global__ __launch_bounds__(512) void pack_weights(
    const float* __restrict__ Wi, const float* __restrict__ Wj,
    __bf16* __restrict__ wiHi, __bf16* __restrict__ wiLo,
    __bf16* __restrict__ wjHi, __bf16* __restrict__ wjLo)
{
    int t = threadIdx.x;                 // 0..511
    int lane = t & 31;
    int ct   = (t >> 5) & 3;
    int kc   = t >> 7;                   // 0..3 for W_i
    int half = lane >> 4;
    int nn   = lane & 15;
    int col  = ct * 16 + nn;

    // W_i fragments (K total = 128)
    #pragma unroll
    for (int e = 0; e < 16; ++e) {
        int K = kc * 32 + half * 16 + e;
        float wv = Wi[K * 64 + col];
        __bf16 hi = (__bf16)wv;
        float  lo = wv - (float)hi;
        wiHi[(size_t)t * 16 + e] = hi;
        wiLo[(size_t)t * 16 + e] = (__bf16)lo;
    }
    // W_j fragments (K total = 64) -> kc in {0,1}: threads 0..255
    if (t < 256) {
        #pragma unroll
        for (int e = 0; e < 16; ++e) {
            int K = kc * 32 + half * 16 + e;
            float wv = Wj[K * 64 + col];
            __bf16 hi = (__bf16)wv;
            float  lo = wv - (float)hi;
            wjHi[(size_t)t * 16 + e] = hi;
            wjLo[(size_t)t * 16 + e] = (__bf16)lo;
        }
    }
}

// ---------------------------------------------------------------------------
// Kernel 2: per-node gated dense via split-bf16 WMMA + segment-sum into pooled
//   block = 256 threads = 8 waves; each wave owns a 16-row tile (128 rows/blk)
// ---------------------------------------------------------------------------
__global__ __launch_bounds__(256) void node_kernel(
    const float* __restrict__ x, const float* __restrict__ h,
    const float* __restrict__ bi, const float* __restrict__ bj,
    const int*   __restrict__ seg,
    const v16bf* __restrict__ wiHi, const v16bf* __restrict__ wiLo,
    const v16bf* __restrict__ wjHi, const v16bf* __restrict__ wjLo,
    float* __restrict__ pooled, int n)
{
    __shared__ float ldsRR[128 * 65];   // padded stride to dodge bank conflicts
    __shared__ int   ldsSeg[128];

    const int t    = threadIdx.x;
    const int lane = t & 31;
    const int w    = t >> 5;            // wave id 0..7
    const int half = lane >> 4;         // 0 / 1
    const int m    = lane & 15;         // row-in-tile (A) / col-in-tile (B,C)

    const int blockBase = blockIdx.x * 128;
    const int tileBase  = blockBase + w * 16;
    const int row       = tileBase + m;
    const int rc        = row < n ? row : (n - 1);
    const float valid   = row < n ? 1.0f : 0.0f;

    if (t < 128) {
        int rr = blockBase + t;
        ldsSeg[t] = seg[rr < n ? rr : (n - 1)];
    }

    v8f ai[4], aj[4];
    #pragma unroll
    for (int ct = 0; ct < 4; ++ct) {
        ai[ct] = (v8f){0.f,0.f,0.f,0.f,0.f,0.f,0.f,0.f};
        aj[ct] = (v8f){0.f,0.f,0.f,0.f,0.f,0.f,0.f,0.f};
    }

    // K chunks of 32: kc 0,1 come from h, kc 2,3 from x (hx = concat(h, x))
    #pragma unroll
    for (int kc = 0; kc < 4; ++kc) {
        const float* src = (kc < 2) ? (h + (size_t)rc * 64 + kc * 32)
                                    : (x + (size_t)rc * 64 + (kc - 2) * 32);
        // A layout (16-bit, 16x32): lane L, elem e -> K = half*16 + e
        const float4* s4 = (const float4*)src;
        float4 p0 = s4[half * 4 + 0];   // K = half*16 .. +3
        float4 p1 = s4[half * 4 + 1];   // K = half*16+4 .. +7
        float4 p2 = s4[half * 4 + 2];   // K = half*16+8 .. +11
        float4 p3 = s4[half * 4 + 3];   // K = half*16+12 .. +15
        float f[16] = { p0.x, p0.y, p0.z, p0.w, p1.x, p1.y, p1.z, p1.w,
                        p2.x, p2.y, p2.z, p2.w, p3.x, p3.y, p3.z, p3.w };
        v16bf ah, al;
        #pragma unroll
        for (int e = 0; e < 16; ++e) {
            float v = f[e] * valid;
            __bf16 hi = (__bf16)v;
            ah[e] = hi;
            al[e] = (__bf16)(v - (float)hi);
        }
        #pragma unroll
        for (int ct = 0; ct < 4; ++ct) {
            v16bf bh = wiHi[(kc * 4 + ct) * 32 + lane];
            v16bf bl = wiLo[(kc * 4 + ct) * 32 + lane];
            ai[ct] = __builtin_amdgcn_wmma_f32_16x16x32_bf16(false, ah, false, bh, (short)0, ai[ct], false, false);
            ai[ct] = __builtin_amdgcn_wmma_f32_16x16x32_bf16(false, ah, false, bl, (short)0, ai[ct], false, false);
            ai[ct] = __builtin_amdgcn_wmma_f32_16x16x32_bf16(false, al, false, bh, (short)0, ai[ct], false, false);
            if (kc < 2) {
                v16bf jh = wjHi[(kc * 4 + ct) * 32 + lane];
                v16bf jl = wjLo[(kc * 4 + ct) * 32 + lane];
                aj[ct] = __builtin_amdgcn_wmma_f32_16x16x32_bf16(false, ah, false, jh, (short)0, aj[ct], false, false);
                aj[ct] = __builtin_amdgcn_wmma_f32_16x16x32_bf16(false, ah, false, jl, (short)0, aj[ct], false, false);
                aj[ct] = __builtin_amdgcn_wmma_f32_16x16x32_bf16(false, al, false, jh, (short)0, aj[ct], false, false);
            }
        }
    }

    // Epilogue: RR = sigmoid(tanh(i)) * relu(j); C layout: M = r + 8*half, N = m
    #pragma unroll
    for (int ct = 0; ct < 4; ++ct) {
        int col = ct * 16 + m;
        float bic = bi[col];
        float bjc = bj[col];
        #pragma unroll
        for (int r = 0; r < 8; ++r) {
            float iacc = ai[ct][r] + bic;
            // tanh via fast exp: tanh(z) = 1 - 2/(e^{2z}+1)
            float e2 = __expf(2.0f * iacc);
            float iv = 1.0f - 2.0f / (e2 + 1.0f);
            float jv = fmaxf(aj[ct][r] + bjc, 0.0f);
            float sg = 1.0f / (1.0f + __expf(-iv));
            float rr = sg * jv * valid;
            int rowLocal = w * 16 + r + half * 8;
            ldsRR[rowLocal * 65 + col] = rr;
        }
    }
    __syncthreads();

    // Segment-run reduction: segment_ids sorted -> runs are contiguous.
    // thread t: col = t&63, rows grp*32 .. grp*32+31
    int col = t & 63;
    int grp = t >> 6;
    int r0  = grp * 32;
    float acc = 0.0f;
    int cur = ldsSeg[r0];
    for (int r = r0; r < r0 + 32; ++r) {
        int s = ldsSeg[r];
        if (s != cur) {
            __hip_atomic_fetch_add(&pooled[(size_t)cur * 64 + col], acc,
                                   __ATOMIC_RELAXED, __HIP_MEMORY_SCOPE_AGENT);
            acc = 0.0f;
            cur = s;
        }
        acc += ldsRR[r * 65 + col];
    }
    __hip_atomic_fetch_add(&pooled[(size_t)cur * 64 + col], acc,
                           __ATOMIC_RELAXED, __HIP_MEMORY_SCOPE_AGENT);
}

// ---------------------------------------------------------------------------
// Kernel 3: per-graph BN + relu(pooled@W_h1 + b_h1) @ W_out + b_out
//   one block per graph, 256 threads = 256 hidden units
// ---------------------------------------------------------------------------
__global__ __launch_bounds__(256) void readout_kernel(
    const float* __restrict__ pooled,
    const float* __restrict__ gamma, const float* __restrict__ beta,
    const float* __restrict__ mmean, const float* __restrict__ mvar,
    const float* __restrict__ Wh1, const float* __restrict__ bh1,
    const float* __restrict__ Wout, const float* __restrict__ bout,
    float* __restrict__ out)
{
    __shared__ float bnp[64];
    __shared__ float red[256];
    int g = blockIdx.x;
    int t = threadIdx.x;
    if (t < 64) {
        float sc = gamma[t] * rsqrtf(mvar[t] + BN_EPS);
        bnp[t] = (pooled[(size_t)g * 64 + t] - mmean[t]) * sc + beta[t];
    }
    __syncthreads();
    float acc = bh1[t];
    #pragma unroll
    for (int k = 0; k < 64; ++k)
        acc = fmaf(bnp[k], Wh1[k * 256 + t], acc);
    float hv = fmaxf(acc, 0.0f);
    red[t] = hv * Wout[t];
    __syncthreads();
    for (int s = 128; s > 0; s >>= 1) {
        if (t < s) red[t] += red[t + s];
        __syncthreads();
    }
    if (t == 0) out[g] = red[0] + bout[0];
}

// ---------------------------------------------------------------------------
extern "C" void kernel_launch(void* const* d_in, const int* in_sizes, int n_in,
                              void* d_out, int out_size, void* d_ws, size_t ws_size,
                              hipStream_t stream) {
    const float* x     = (const float*)d_in[0];
    const float* h     = (const float*)d_in[1];
    const float* W_i   = (const float*)d_in[2];
    const float* b_i   = (const float*)d_in[3];
    const float* W_j   = (const float*)d_in[4];
    const float* b_j   = (const float*)d_in[5];
    const float* gamma = (const float*)d_in[6];
    const float* beta  = (const float*)d_in[7];
    const float* mmean = (const float*)d_in[8];
    const float* mvar  = (const float*)d_in[9];
    const float* W_h1  = (const float*)d_in[10];
    const float* b_h1  = (const float*)d_in[11];
    const float* W_out = (const float*)d_in[12];
    const float* b_out = (const float*)d_in[13];
    const int*   seg   = (const int*)d_in[14];

    const int n = in_sizes[0] / 64;   // N nodes
    const int G = out_size;           // graphs ([G,1] output)

    char* ws = (char*)d_ws;
    float* pooled = (float*)ws;
    size_t off = (size_t)G * 64 * sizeof(float);
    __bf16* wiHi = (__bf16*)(ws + off); off += 512 * 16 * sizeof(__bf16);
    __bf16* wiLo = (__bf16*)(ws + off); off += 512 * 16 * sizeof(__bf16);
    __bf16* wjHi = (__bf16*)(ws + off); off += 256 * 16 * sizeof(__bf16);
    __bf16* wjLo = (__bf16*)(ws + off);

    zero_pooled<<<(G * 64 + 255) / 256, 256, 0, stream>>>(pooled, G * 64);
    pack_weights<<<1, 512, 0, stream>>>(W_i, W_j, wiHi, wiLo, wjHi, wjLo);
    node_kernel<<<(n + 127) / 128, 256, 0, stream>>>(
        x, h, b_i, b_j, seg,
        (const v16bf*)wiHi, (const v16bf*)wiLo,
        (const v16bf*)wjHi, (const v16bf*)wjLo,
        pooled, n);
    readout_kernel<<<G, 256, 0, stream>>>(pooled, gamma, beta, mmean, mvar,
                                          W_h1, b_h1, W_out, b_out, (float*)d_out);
}
